// ScaledDotProductAttention_20040317403737
// MI455X (gfx1250) — compile-verified, matrix operand order checked by reading, not run on hardware
//
#include <hip/hip_runtime.h>
#include <hip/hip_bf16.h>
#include <math.h>

typedef __attribute__((ext_vector_type(16))) __bf16 v16bf;
typedef __attribute__((ext_vector_type(8)))  float  v8f;

constexpr int B = 4, H = 16, S = 2048, D = 128;
constexpr int SROW   = S + 4;   // fp32 score row pitch (pad: kills 2-way bank conflict between lane halves)
constexpr int QPITCH = D + 8;   // bf16 Q row pitch
constexpr float SCALE = 0.08838834764831845f; // 1/sqrt(128)

static __device__ __forceinline__ __bf16 f2bf(float x) { return (__bf16)x; }

__global__ __launch_bounds__(256, 1)
void attn_fwd_causal_kernel(const float* __restrict__ Q,
                            const float* __restrict__ K,
                            const float* __restrict__ V,
                            float* __restrict__ Out,
                            float* __restrict__ Wts)
{
    __shared__ float  sc[16 * SROW];     // 16 x 2052 f32  (~128.2 KB) scores -> probabilities
    __shared__ __bf16 qs[16 * QPITCH];   // 16 x 136 bf16  (~4.25 KB)  Q tile

    const int qt = blockIdx.x;                 // query tile: 0..127
    const int h  = blockIdx.y;
    const int b  = blockIdx.z;
    const int q0 = qt * 16;

    const size_t bh   = (size_t)b * H + h;
    const size_t base = bh * (size_t)S * D;    // element base into Q/K/V for this (b,h)

    const int tid  = threadIdx.x;
    const int wave = tid >> 5;                 // 0..7 (wave32)
    const int lane = tid & 31;
    const int half = lane >> 4;                // lane half selects K-subblock per ISA layout
    const int l16  = lane & 15;

    // ---------- stage Q tile (16 x 128 fp32 -> bf16) in LDS ----------
    #pragma unroll
    for (int e = 0; e < 8; ++e) {              // 2048 elems / 256 threads
        int idx = tid + e * 256;
        int r = idx >> 7, c = idx & 127;
        qs[r * QPITCH + c] = f2bf(Q[base + (size_t)(q0 + r) * D + c]);
    }
    __syncthreads();

    // ---------- build the 4 Q A-fragments (16x32 bf16 each) in registers ----------
    // 16-bit A layout: lane = M row (l16); VGPR i holds K = (i&3)*2 + (i>>2)*16 + half*8, +1
    v16bf qfrag[4];
    #pragma unroll
    for (int c = 0; c < 4; ++c) {
        #pragma unroll
        for (int i = 0; i < 8; ++i) {
            int k = c * 32 + (i & 3) * 2 + (i >> 2) * 16 + half * 8;
            qfrag[c][2 * i]     = qs[l16 * QPITCH + k];
            qfrag[c][2 * i + 1] = qs[l16 * QPITCH + k + 1];
        }
    }

    // ---------- pass 1: S = (Q K^T) * scale for causal tiles kt <= qt ----------
    for (int kt = wave; kt <= qt; kt += 8) {
        const float* kp = K + base + (size_t)(kt * 16 + l16) * D; // lane = key row of this tile
        v8f acc = {};
        #pragma unroll
        for (int c = 0; c < 4; ++c) {
            v16bf bfrag;                      // B = K^T  => fragment == A-layout of K rows
            #pragma unroll
            for (int i = 0; i < 8; ++i) {
                int k = c * 32 + (i & 3) * 2 + (i >> 2) * 16 + half * 8;
                bfrag[2 * i]     = f2bf(kp[k]);
                bfrag[2 * i + 1] = f2bf(kp[k + 1]);
            }
            acc = __builtin_amdgcn_wmma_f32_16x16x32_bf16(false, qfrag[c],
                                                          false, bfrag,
                                                          (short)0, acc, false, false);
        }
        // C/D layout: VGPR r -> M = r + 8*half, N = l16
        #pragma unroll
        for (int r = 0; r < 8; ++r) {
            int m = r + 8 * half;
            sc[m * SROW + kt * 16 + l16] = acc[r] * SCALE;
        }
    }
    __syncthreads();

    // ---------- softmax (2 rows per wave), stream weights to global (NT) ----------
    const float* dummy = nullptr; (void)dummy;
    #pragma unroll
    for (int rr = 0; rr < 2; ++rr) {
        int m = wave * 2 + rr;
        int q = q0 + m;                        // causal bound: keys k <= q are live
        float* srow = &sc[m * SROW];

        float mx = -INFINITY;
        for (int k = lane; k <= q; k += 32) mx = fmaxf(mx, srow[k]);
        #pragma unroll
        for (int o = 16; o >= 1; o >>= 1) mx = fmaxf(mx, __shfl_xor(mx, o, 32));

        float sum = 0.f;
        for (int k = lane; k <= q; k += 32) {
            float e = __expf(srow[k] - mx);
            srow[k] = e;
            sum += e;
        }
        #pragma unroll
        for (int o = 16; o >= 1; o >>= 1) sum += __shfl_xor(sum, o, 32);
        float inv = 1.f / sum;

        float* wrow = Wts + (bh * S + (size_t)q) * S;
        for (int k = lane; k < S; k += 32) {   // full row: normalize live part, zero masked tail
            float p = (k <= q) ? srow[k] * inv : 0.f;
            srow[k] = p;                       // leave P in LDS for the P*V matmul
            __builtin_nontemporal_store(p, wrow + k);  // 1 GB write-once stream: bypass L2 retention
        }
    }
    __syncthreads();

    // ---------- O = P * V ; wave owns output columns [16*wave, 16*wave+16) ----------
    const int col = wave * 16 + l16;
    const float* vp = V + base + col;
    v8f oacc = {};
    const int nchunk = (q0 + 47) >> 5;         // 32-wide K chunks covering k <= q0+15
    for (int c2 = 0; c2 < nchunk; ++c2) {
        v16bf afrag, bfrag;
        #pragma unroll
        for (int i = 0; i < 8; ++i) {
            int k = c2 * 32 + (i & 3) * 2 + (i >> 2) * 16 + half * 8;
            afrag[2 * i]     = f2bf(sc[l16 * SROW + k]);
            afrag[2 * i + 1] = f2bf(sc[l16 * SROW + k + 1]);
            bfrag[2 * i]     = f2bf(vp[(size_t)k * D]);        // B = V: lane = N column
            bfrag[2 * i + 1] = f2bf(vp[(size_t)(k + 1) * D]);
        }
        oacc = __builtin_amdgcn_wmma_f32_16x16x32_bf16(false, afrag,
                                                       false, bfrag,
                                                       (short)0, oacc, false, false);
    }
    #pragma unroll
    for (int r = 0; r < 8; ++r) {
        int m = r + 8 * half;
        Out[base + (size_t)(q0 + m) * D + col] = oacc[r];
    }
}

extern "C" void kernel_launch(void* const* d_in, const int* in_sizes, int n_in,
                              void* d_out, int out_size, void* d_ws, size_t ws_size,
                              hipStream_t stream) {
    (void)in_sizes; (void)n_in; (void)out_size; (void)d_ws; (void)ws_size;
    const float* Qp = (const float*)d_in[0];
    const float* Kp = (const float*)d_in[1];
    const float* Vp = (const float*)d_in[2];
    // d_in[3] is the causal tril mask; causality is baked into the kernel's loop bounds.
    float* out = (float*)d_out;                              // (B,H,S,D) first
    float* wts = out + (size_t)B * H * S * D;                // then (B,H,S,S) weights
    attn_fwd_causal_kernel<<<dim3(S / 16, H, B), 256, 0, stream>>>(Qp, Kp, Vp, out, wts);
}
